// MEVGraphSAGE_53996328846126
// MI455X (gfx1250) — matrix operand order repaired and checked by reading, not compiled, not run on hardware
//
#include <hip/hip_runtime.h>
#include <hip/hip_bf16.h>
#include <stdint.h>

// ---------------- model constants ----------------
#define N_NODES   50000
#define N_EDGES   800000
#define IN_CH     128
#define HID       256
#define NUM_GRAPHS 64
#define MLP1      256
#define MLP2      128
#define OUT_CH    2
#define BN_EPS    1e-5f
#define INV_TEMP  (1.0f / 1.5f)

typedef __bf16 bf16;
typedef bf16  v16bf __attribute__((ext_vector_type(16)));
typedef bf16  v8bf  __attribute__((ext_vector_type(8)));
typedef float v8f   __attribute__((ext_vector_type(8)));
typedef unsigned int u32x4 __attribute__((ext_vector_type(4)));
typedef int          i32x8 __attribute__((ext_vector_type(8)));
typedef int          i32x4 __attribute__((ext_vector_type(4)));

struct __align__(8) bf4 { bf16 v[4]; };

// ---------------- small utility kernels ----------------
__global__ void zero_f(float* p, int n) {
    int i = blockIdx.x * blockDim.x + threadIdx.x;
    if (i < n) p[i] = 0.0f;
}

__global__ void cvt_f32_bf16(const float* __restrict__ in, bf16* __restrict__ out, int n) {
    int i = blockIdx.x * blockDim.x + threadIdx.x;
    if (i < n) out[i] = (bf16)in[i];
}

// ---------------- degree ----------------
__global__ void degree_kernel(const int* __restrict__ dst, float* __restrict__ deg) {
    int e = blockIdx.x * blockDim.x + threadIdx.x;
    if (e < N_EDGES) atomicAdd(&deg[dst[e]], 1.0f);
}

__global__ void inv_deg_kernel(float* deg) {
    int i = blockIdx.x * blockDim.x + threadIdx.x;
    if (i < N_NODES) {
        float d = deg[i];
        deg[i] = 1.0f / fmaxf(d, 1.0f);
    }
}

// ---------------- input batch-norm ----------------
// stats: [0:128)=sum, [128:256)=sumsq, [256:384)=scale, [384:512)=shift
__global__ void bn_stats_kernel(const float* __restrict__ x, float* __restrict__ stats) {
    const int ROWS = 250;                     // 200 blocks * 250 rows = 50000
    int c = threadIdx.x;                      // 128 threads
    int r0 = blockIdx.x * ROWS;
    float s = 0.0f, s2 = 0.0f;
    for (int r = 0; r < ROWS; ++r) {
        float v = x[(size_t)(r0 + r) * IN_CH + c];
        s += v; s2 += v * v;
    }
    atomicAdd(&stats[c], s);
    atomicAdd(&stats[IN_CH + c], s2);
}

__global__ void bn_finalize_kernel(const float* __restrict__ g, const float* __restrict__ b,
                                   float* __restrict__ stats) {
    int c = threadIdx.x;                      // 128 threads
    float m = stats[c] * (1.0f / N_NODES);
    float v = stats[IN_CH + c] * (1.0f / N_NODES) - m * m;
    float sc = rsqrtf(v + BN_EPS) * g[c];
    stats[2 * IN_CH + c] = sc;
    stats[3 * IN_CH + c] = b[c] - m * sc;
}

__global__ void bn_apply_kernel(const float* __restrict__ x, const float* __restrict__ stats,
                                bf16* __restrict__ h0) {
    int i = blockIdx.x * blockDim.x + threadIdx.x;
    if (i < N_NODES * IN_CH) {
        int c = i & (IN_CH - 1);
        h0[i] = (bf16)(x[i] * stats[2 * IN_CH + c] + stats[3 * IN_CH + c]);
    }
}

// ---------------- edge scatter-add (segment_sum of h[src] at dst) ----------------
__global__ void scatter_add_kernel(const int* __restrict__ src, const int* __restrict__ dst,
                                   const bf16* __restrict__ H, float* __restrict__ agg,
                                   int K, int total) {
    int idx = blockIdx.x * blockDim.x + threadIdx.x;
    if (idx >= total) return;
    int chunks = K >> 2;
    int e = idx / chunks;
    int c = (idx - e * chunks) << 2;
    int s = src[e], d = dst[e];
    bf4 hv = *(const bf4*)(H + (size_t)s * K + c);
    float* ap = agg + (size_t)d * K + c;
    atomicAdd(ap + 0, (float)hv.v[0]);
    atomicAdd(ap + 1, (float)hv.v[1]);
    atomicAdd(ap + 2, (float)hv.v[2]);
    atomicAdd(ap + 3, (float)hv.v[3]);
}

// agg_bf16 = bf16(agg_f32 * inv_deg[row])
__global__ void scale_cvt_kernel(const float* __restrict__ aggF, const float* __restrict__ inv_deg,
                                 bf16* __restrict__ aggB, int K, int total) {
    int i = blockIdx.x * blockDim.x + threadIdx.x;
    if (i < total) {
        int row = i / K;
        aggB[i] = (bf16)(aggF[i] * inv_deg[row]);
    }
}

// ---------------- TDM: DMA a [rows x K] bf16 tile (row-major, contiguous) into LDS ----------
// Builds a Tensor DMA Descriptor per CDNA5 ISA 8.3/8.4 and issues tensor_load_to_lds.
// LDS padding: pad_amount = 4 DWORDs (16B) after every full row (pad_interval = row DWORDs),
// giving an LDS row stride of K*2+16 bytes (16*odd -> conflict-free b128 fragment loads).
__device__ __forceinline__ void tdm_load_tile(const bf16* gsrc, uint32_t lds_off, int K, int rows) {
    uint64_t ga = (uint64_t)(uintptr_t)gsrc;
    u32x4 g0;
    g0[0] = 1u;                                       // count=1, user descriptor
    g0[1] = lds_off;                                  // lds_addr (bytes)
    g0[2] = (uint32_t)ga;                             // global_addr[31:0]
    g0[3] = (uint32_t)((ga >> 32) & 0x01FFFFFFu) | (2u << 30);   // addr[56:32] | type=2
    // pad_interval: bytesStored>>3 >= (1<<c)  -> row of K*2 bytes: c = log2(K*2/8)
    uint32_t pad_int = (K == 256) ? 6u : 5u;          // 512B row -> 6, 256B row -> 5
    uint32_t w0 = (1u << 16)                          // data_size = 2 bytes
                | (1u << 20)                          // pad_enable
                | (pad_int << 22)                     // pad_interval
                | (3u << 25);                         // pad_amount = 4 DWORDs (16B)
    uint32_t td0 = (uint32_t)K;                       // tensor_dim0 (elements)
    uint32_t td1 = (uint32_t)rows;                    // tensor_dim1 (rows, rel. tile start)
    i32x8 g1;
    g1[0] = (int)w0;
    g1[1] = (int)((td0 & 0xFFFFu) << 16);                       // dim0[15:0] at bits 63:48
    g1[2] = (int)((td0 >> 16) | ((td1 & 0xFFFFu) << 16));       // dim0[31:16] | dim1[15:0]
    g1[3] = (int)((td1 >> 16) | ((uint32_t)K << 16));           // dim1[31:16] | tile_dim0=K
    g1[4] = (int)(uint32_t)rows;                                // tile_dim1=rows, tile_dim2=0
    g1[5] = (int)(uint32_t)K;                                   // tensor_dim0_stride low32
    g1[6] = 0;                                                  // stride hi / dim1_stride lo
    g1[7] = 0;
    i32x4 z4 = {0, 0, 0, 0};                                    // groups 2/3 unused (2D)
    i32x8 z8 = {0, 0, 0, 0, 0, 0, 0, 0};                        // extra group (unused)
    __builtin_amdgcn_tensor_load_to_lds(g0, g1, z4, z4, z8, 0); // 6-arg toolchain form
}

// ---------------- fused SAGE GEMM: Hout = relu(Aagg@Wl^T + Ah@Wr^T + bias) ----------------
// One block (512 thr = 16 waves) per 16-row M tile; wave w owns N-tile w (O = 256 = 16 tiles).
// The two 16xK A tiles are DMA'd into LDS once by the TDM and shared by all 16 waves.
// Fragment layouts per CDNA5 ISA 7.12.2 (16-bit A 16x32; B 32x16; 32-bit C 16x16).
#define LDS_PAD   16                                  // bytes per row (from TDM pad)
#define LDS_SROW  (HID * 2 + LDS_PAD)                 // max row stride (K=256): 528B
__device__ __forceinline__ v16bf lds_a_frag(const char* base, int srow, int m, int k0, int hh) {
    const char* rp = base + m * srow + k0 * 2;
    v8bf lo = *(const v8bf*)(rp + 32 * hh);           // K = k0 + 16h .. +7
    v8bf hi = *(const v8bf*)(rp + 16 + 32 * hh);      // K = k0 + 16h + 8 .. +15
    v16bf a;
#pragma unroll
    for (int i = 0; i < 8; ++i) { a[i] = lo[i]; a[8 + i] = hi[i]; }
    return a;
}

__global__ __launch_bounds__(512)
void sage_gemm_kernel(const bf16* __restrict__ Aagg, const bf16* __restrict__ Ah,
                      const bf16* __restrict__ Wl,  const bf16* __restrict__ Wr,
                      const float* __restrict__ bias, bf16* __restrict__ Hout,
                      int K, int O) {
    __shared__ __align__(16) char lds_buf[2 * 16 * LDS_SROW];
    const int lane  = threadIdx.x & 31;
    const int wave  = threadIdx.x >> 5;               // 0..15 -> N tile
    const int m     = lane & 15;
    const int hh    = lane >> 4;
    const int mbase = blockIdx.x << 4;
    const int srow  = K * 2 + LDS_PAD;                // actual padded LDS row stride

    if (wave == 0) {                                  // one TDM issue per block
        uint32_t lds0 = (uint32_t)(uintptr_t)&lds_buf[0];       // LDS offset = addr[31:0]
        tdm_load_tile(Aagg + (size_t)mbase * K, lds0,            K, 16);
        tdm_load_tile(Ah   + (size_t)mbase * K, lds0 + 16 * srow, K, 16);
        __builtin_amdgcn_s_wait_tensorcnt(0);
    }
    __syncthreads();                                  // publish LDS tiles to all 16 waves

    const char* ldsA = lds_buf;
    const char* ldsH = lds_buf + 16 * srow;
    const int ncol   = (wave << 4) + m;               // output channel (B row / C column)
    const size_t rowW = (size_t)ncol * K;

    v8f acc = {};
    for (int k0 = 0; k0 < K; k0 += 32) {
        if (k0 + 32 < K) {
            __builtin_prefetch(Wl + rowW + k0 + 32, 0, 3);
            __builtin_prefetch(Wr + rowW + k0 + 32, 0, 3);
        }
        v16bf a1 = lds_a_frag(ldsA, srow, m, k0, hh);
        v16bf b1 = *(const v16bf*)(Wl + rowW + k0 + 16 * hh);   // K = k0+16h .. +15
        acc = __builtin_amdgcn_wmma_f32_16x16x32_bf16(false, a1, false, b1,
                                                      (short)0, acc, false, false);
        v16bf a2 = lds_a_frag(ldsH, srow, m, k0, hh);
        v16bf b2 = *(const v16bf*)(Wr + rowW + k0 + 16 * hh);
        acc = __builtin_amdgcn_wmma_f32_16x16x32_bf16(false, a2, false, b2,
                                                      (short)0, acc, false, false);
    }

    const float bv = bias[ncol];
#pragma unroll
    for (int r = 0; r < 8; ++r) {                     // C: reg r -> row r + 8h, col = lane&15
        float v = acc[r] + bv;
        v = v > 0.0f ? v : 0.0f;                      // ReLU
        Hout[(size_t)(mbase + (hh << 3) + r) * O + ncol] = (bf16)v;
    }
}

// ---------------- graph pooling ----------------
__global__ void counts_kernel(const int* __restrict__ batch, float* __restrict__ counts) {
    int n = blockIdx.x * blockDim.x + threadIdx.x;
    if (n < N_NODES) atomicAdd(&counts[batch[n]], 1.0f);
}

__global__ void pool_kernel(const bf16* __restrict__ H, const int* __restrict__ batch,
                            float* __restrict__ psum, float* __restrict__ pmax) {
    int idx = blockIdx.x * blockDim.x + threadIdx.x;
    if (idx >= N_NODES * HID) return;
    int n = idx >> 8;                 // /HID
    int c = idx & (HID - 1);
    int g = batch[n];
    float v = (float)H[idx];
    atomicAdd(&psum[g * HID + c], v);
    // post-ReLU values are >= 0 and pmax is init'd to +0.0, so uint max == float max
    atomicMax((unsigned int*)&pmax[g * HID + c], __float_as_uint(v));
}

// pooled batch-norm over G=64 rows, C=512 channels (mean||max), single block of 512 threads
__global__ void pool_bn_kernel(const float* __restrict__ psum, const float* __restrict__ pmax,
                               const float* __restrict__ counts,
                               const float* __restrict__ pg, const float* __restrict__ pb,
                               float* __restrict__ pooled) {
    int c = threadIdx.x;              // 0..511
    float s = 0.0f;
    for (int g = 0; g < NUM_GRAPHS; ++g) {
        float v = (c < HID) ? psum[g * HID + c] / fmaxf(counts[g], 1.0f)
                            : pmax[g * HID + (c - HID)];
        s += v;
    }
    float mn = s * (1.0f / NUM_GRAPHS);
    float s2 = 0.0f;
    for (int g = 0; g < NUM_GRAPHS; ++g) {
        float v = (c < HID) ? psum[g * HID + c] / fmaxf(counts[g], 1.0f)
                            : pmax[g * HID + (c - HID)];
        float d = v - mn;
        s2 += d * d;
    }
    float inv = rsqrtf(s2 * (1.0f / NUM_GRAPHS) + BN_EPS);
    for (int g = 0; g < NUM_GRAPHS; ++g) {
        float v = (c < HID) ? psum[g * HID + c] / fmaxf(counts[g], 1.0f)
                            : pmax[g * HID + (c - HID)];
        pooled[g * (2 * HID) + c] = (v - mn) * inv * pg[c] + pb[c];
    }
}

// ---------------- MLP head (fp32, tiny) ----------------
// mode: 1 = relu, 2 = * (1/TEMPERATURE)
__global__ void fc_kernel(const float* __restrict__ in, const float* __restrict__ W,
                          const float* __restrict__ b, float* __restrict__ out,
                          int G, int Kdim, int O, int mode) {
    int idx = blockIdx.x * blockDim.x + threadIdx.x;
    if (idx >= G * O) return;
    int g = idx / O;
    int o = idx - g * O;
    float s = b[o];
    const float* ip = in + (size_t)g * Kdim;
    const float* wp = W + (size_t)o * Kdim;
    for (int k = 0; k < Kdim; ++k) s += ip[k] * wp[k];
    if (mode == 1) s = fmaxf(s, 0.0f);
    else if (mode == 2) s *= INV_TEMP;
    out[idx] = s;
}

// ---------------- host side ----------------
static inline int cdiv(int a, int b) { return (a + b - 1) / b; }

extern "C" void kernel_launch(void* const* d_in, const int* in_sizes, int n_in,
                              void* d_out, int out_size, void* d_ws, size_t ws_size,
                              hipStream_t stream) {
    (void)in_sizes; (void)n_in; (void)out_size; (void)ws_size;

    // -------- inputs (setup_inputs order) --------
    const float* x        = (const float*)d_in[0];
    const int*   eidx     = (const int*)  d_in[1];
    const int*   batch    = (const int*)  d_in[2];
    const float* bn_g     = (const float*)d_in[3];
    const float* bn_b     = (const float*)d_in[4];
    const float* wl_f[4]  = {(const float*)d_in[5],  (const float*)d_in[8],
                             (const float*)d_in[11], (const float*)d_in[14]};
    const float* bs_f[4]  = {(const float*)d_in[6],  (const float*)d_in[9],
                             (const float*)d_in[12], (const float*)d_in[15]};
    const float* wr_f[4]  = {(const float*)d_in[7],  (const float*)d_in[10],
                             (const float*)d_in[13], (const float*)d_in[16]};
    const float* pbn_g    = (const float*)d_in[17];
    const float* pbn_b    = (const float*)d_in[18];
    const float* wl1      = (const float*)d_in[19];
    const float* bl1      = (const float*)d_in[20];
    const float* wl2      = (const float*)d_in[21];
    const float* bl2      = (const float*)d_in[22];
    const float* wl3      = (const float*)d_in[23];
    const float* bl3      = (const float*)d_in[24];
    float* out            = (float*)d_out;

    const int* esrc = eidx;
    const int* edst = eidx + N_EDGES;

    // -------- workspace bump allocator --------
    char* wp = (char*)d_ws;
    auto alloc = [&](size_t bytes) -> void* {
        void* r = (void*)wp;
        wp += (bytes + 255) & ~(size_t)255;
        return r;
    };
    float* deg    = (float*)alloc((size_t)N_NODES * 4);
    float* stats  = (float*)alloc(4 * IN_CH * 4);            // sum|sumsq|scale|shift
    float* counts = (float*)alloc(NUM_GRAPHS * 4);
    float* psum   = (float*)alloc((size_t)NUM_GRAPHS * HID * 4);
    float* pmax   = (float*)alloc((size_t)NUM_GRAPHS * HID * 4);
    float* pooled = (float*)alloc((size_t)NUM_GRAPHS * 2 * HID * 4);
    float* z1     = (float*)alloc((size_t)NUM_GRAPHS * MLP1 * 4);
    float* z2     = (float*)alloc((size_t)NUM_GRAPHS * MLP2 * 4);
    bf16* h0      = (bf16*)alloc((size_t)N_NODES * IN_CH * 2);
    bf16* hA      = (bf16*)alloc((size_t)N_NODES * HID * 2);
    bf16* hB      = (bf16*)alloc((size_t)N_NODES * HID * 2);
    float* aggF   = (float*)alloc((size_t)N_NODES * HID * 4);
    bf16* aggB    = (bf16*)alloc((size_t)N_NODES * HID * 2);
    bf16* wl_b[4]; bf16* wr_b[4];
    const int wdims[4] = {HID * IN_CH, HID * HID, HID * HID, HID * HID};
    for (int l = 0; l < 4; ++l) {
        wl_b[l] = (bf16*)alloc((size_t)wdims[l] * 2);
        wr_b[l] = (bf16*)alloc((size_t)wdims[l] * 2);
    }

    const int T = 256;

    // -------- zero scratch --------
    zero_f<<<cdiv(N_NODES, T), T, 0, stream>>>(deg, N_NODES);
    zero_f<<<cdiv(4 * IN_CH, T), T, 0, stream>>>(stats, 4 * IN_CH);
    zero_f<<<1, NUM_GRAPHS, 0, stream>>>(counts, NUM_GRAPHS);
    zero_f<<<cdiv(NUM_GRAPHS * HID, T), T, 0, stream>>>(psum, NUM_GRAPHS * HID);
    zero_f<<<cdiv(NUM_GRAPHS * HID, T), T, 0, stream>>>(pmax, NUM_GRAPHS * HID);

    // -------- degree / inv-degree --------
    degree_kernel<<<cdiv(N_EDGES, T), T, 0, stream>>>(edst, deg);
    inv_deg_kernel<<<cdiv(N_NODES, T), T, 0, stream>>>(deg);

    // -------- input batch-norm -> h0 (bf16) --------
    bn_stats_kernel<<<200, IN_CH, 0, stream>>>(x, stats);
    bn_finalize_kernel<<<1, IN_CH, 0, stream>>>(bn_g, bn_b, stats);
    bn_apply_kernel<<<cdiv(N_NODES * IN_CH, T), T, 0, stream>>>(x, stats, h0);

    // -------- weights -> bf16 --------
    for (int l = 0; l < 4; ++l) {
        cvt_f32_bf16<<<cdiv(wdims[l], T), T, 0, stream>>>(wl_f[l], wl_b[l], wdims[l]);
        cvt_f32_bf16<<<cdiv(wdims[l], T), T, 0, stream>>>(wr_f[l], wr_b[l], wdims[l]);
    }

    // -------- 4 SAGE layers --------
    const bf16* hin[4]  = {h0, hA, hB, hA};
    bf16*       hout[4] = {hA, hB, hA, hB};
    const int   Ks[4]   = {IN_CH, HID, HID, HID};
    for (int l = 0; l < 4; ++l) {
        const int K = Ks[l];
        const int nagg = N_NODES * K;
        zero_f<<<cdiv(nagg, T), T, 0, stream>>>(aggF, nagg);
        const int total = N_EDGES * (K >> 2);
        scatter_add_kernel<<<cdiv(total, T), T, 0, stream>>>(esrc, edst, hin[l], aggF, K, total);
        scale_cvt_kernel<<<cdiv(nagg, T), T, 0, stream>>>(aggF, deg, aggB, K, nagg);
        // one block per 16-row M tile; 16 waves cover the 16 N tiles of O=256
        sage_gemm_kernel<<<N_NODES / 16, 512, 0, stream>>>(aggB, hin[l], wl_b[l], wr_b[l],
                                                           bs_f[l], hout[l], K, HID);
    }
    const bf16* hfin = hout[3];

    // -------- pooling --------
    counts_kernel<<<cdiv(N_NODES, T), T, 0, stream>>>(batch, counts);
    pool_kernel<<<cdiv(N_NODES * HID, T), T, 0, stream>>>(hfin, batch, psum, pmax);
    pool_bn_kernel<<<1, 2 * HID, 0, stream>>>(psum, pmax, counts, pbn_g, pbn_b, pooled);

    // -------- MLP head --------
    fc_kernel<<<cdiv(NUM_GRAPHS * MLP1, T), T, 0, stream>>>(pooled, wl1, bl1, z1,
                                                            NUM_GRAPHS, 2 * HID, MLP1, 1);
    fc_kernel<<<cdiv(NUM_GRAPHS * MLP2, T), T, 0, stream>>>(z1, wl2, bl2, z2,
                                                            NUM_GRAPHS, MLP1, MLP2, 1);
    fc_kernel<<<1, NUM_GRAPHS * OUT_CH, 0, stream>>>(z2, wl3, bl3, out,
                                                     NUM_GRAPHS, MLP2, OUT_CH, 2);
}
